// DendSeqNet2_45930380263773
// MI455X (gfx1250) — matrix operand order repaired and checked by reading, not compiled, not run on hardware
//
#include <hip/hip_runtime.h>

// ---------------------------------------------------------------------------
// DendSeqNet2 on MI455X (gfx1250): persistent batch-tiled LIF scan with
// bf16 WMMA (v_wmma_f32_16x16x32_bf16), state held in accumulator VGPRs.
// Round 3: keep round-2 pipelining (double-buffered fragments, immediate
// offsets, dual accumulator chains, x[t+1] prefetch) but compute A/B
// addresses inline so clang keeps LDS/global address spaces (ds_load_b128 +
// global_load_b128 instead of flat_load_b128, single-counter waits).
// ---------------------------------------------------------------------------

typedef __attribute__((ext_vector_type(16))) __bf16 v16bf;
typedef __attribute__((ext_vector_type(8)))  float  v8f;

union Frag { v16bf v; uint4 u[2]; };

#define T_STEPS 200
#define BATCH   256
#define F2      784   // 28*28
#define HC      2
#define H1      200
#define SPL1    392
#define OC      4
#define NOUT    10
#define SPL2    50

#define KT_H    13    // ceil(392/32) K-steps, hidden
#define KPAD_H  416   // 13*32
#define NPAD_H  208   // 13*16
#define KT_O    2     // ceil(50/32) K-steps, output
#define NPAD_O  16

#define MB      16    // batch rows per workgroup

#define DVM 0.1f      // DT * TAU_MEM_INV
#define DVS 0.8f      // 1 - DT * TAU_SYN_INV

// --- pack Wh (f32 [2][200][392]) -> Wb bf16 [2][13][208][32], zero padded ---
__global__ void prep_wh_kernel(const float* __restrict__ Wh, __bf16* __restrict__ Wb) {
    int e = blockIdx.x * 256 + threadIdx.x;
    const int TOT = HC * KT_H * NPAD_H * 32;
    if (e >= TOT) return;
    int kk = e & 31;
    int n  = (e >> 5) % NPAD_H;
    int kt = ((e >> 5) / NPAD_H) % KT_H;
    int c  = e / (32 * NPAD_H * KT_H);
    int k  = kt * 32 + kk;
    float v = 0.f;
    if (n < H1 && k < SPL1) v = Wh[((size_t)c * H1 + n) * SPL1 + k];
    Wb[e] = (__bf16)v;
}

// --- pack Wo (f32 [4][10][50]) -> Wob bf16 [4][2][16][32], zero padded ---
__global__ void prep_wo_kernel(const float* __restrict__ Wo, __bf16* __restrict__ Wob) {
    int e = blockIdx.x * 256 + threadIdx.x;
    const int TOT = OC * KT_O * NPAD_O * 32;
    if (e >= TOT) return;
    int kk = e & 31;
    int n  = (e >> 5) % NPAD_O;
    int kt = ((e >> 5) / NPAD_O) % KT_O;
    int c  = e / (32 * NPAD_O * KT_O);
    int k  = kt * 32 + kk;
    float v = 0.f;
    if (n < NOUT && k < SPL2) v = Wo[((size_t)c * NOUT + n) * SPL2 + k];
    Wob[e] = (__bf16)v;
}

// ---------------------------------------------------------------------------
// Main persistent kernel: 16 blocks x 256 threads (8 wave32).
// Block b owns batch rows [16b, 16b+16) for all 200 timesteps; LIF/LI state
// lives in WMMA accumulator-layout VGPRs for the whole scan.
// ---------------------------------------------------------------------------
__global__ void __launch_bounds__(256, 1) dendnet_kernel(
    const float* __restrict__ x,     // [T,256,784]
    const float* __restrict__ bh,    // [2,200]
    const float* __restrict__ bo,    // [4,10]
    const __bf16* __restrict__ Wb,   // [2][13][208][32]
    const __bf16* __restrict__ Wob,  // [4][2][16][32]
    float* __restrict__ out)         // [T,256,10]
{
    __shared__ __attribute__((aligned(16))) __bf16 lx[HC][MB][KPAD_H];  // x tile
    __shared__ __attribute__((aligned(16))) __bf16 lz[HC][MB][NPAD_H];  // spikes
    __shared__ __attribute__((aligned(16))) float  lvo[OC][MB][NPAD_O]; // out volt

    const int tid  = threadIdx.x;
    const int w    = tid >> 5;        // wave id (0..7)
    const int lane = tid & 31;
    const int g    = (lane >> 4) & 1; // lane half
    const int nl   = lane & 15;       // lane-within-half
    const int b0   = blockIdx.x * MB;

    // zero x K-pad columns [392,416) once
    for (int i = tid; i < HC * MB * (KPAD_H - SPL1); i += 256) {
        int col = SPL1 + (i % (KPAD_H - SPL1));
        int m   = (i / (KPAD_H - SPL1)) % MB;
        int c   = i / ((KPAD_H - SPL1) * MB);
        lx[c][m][col] = (__bf16)0.f;
    }

    // hidden tile ownership + persistent LIF state (26 tiles over 8 waves)
    v8f vh[4], ih[4];
    float bhr[4];
    int ntile = 0;
    for (int it = 0; it < 4; ++it) {
        int tt = w + 8 * it;
        if (tt < HC * KT_H) {
            ntile = it + 1;
            for (int r = 0; r < 8; ++r) { vh[it][r] = 0.f; ih[it][r] = 0.f; }
            int c = tt / KT_H, nt = tt % KT_H;
            int n = nt * 16 + nl;
            bhr[it] = (n < H1) ? bh[c * H1 + n] : 0.f;
        }
    }
    // output LI state (waves 0..3, channel = w)
    v8f vo, io;
    for (int r = 0; r < 8; ++r) { vo[r] = 0.f; io[r] = 0.f; }
    float bor = 0.f;
    if (w < OC) bor = (nl < NOUT) ? bo[w * NOUT + nl] : 0.f;

    for (int t = 0; t < T_STEPS; ++t) {
        // ---- stage x[t] batch tile into LDS as bf16; prefetch x[t+1] ----
        const float* xt = x + ((size_t)t * BATCH + b0) * F2;
        for (int i = tid; i < MB * (F2 / 4); i += 256) {
            int m   = i / (F2 / 4);
            int q   = i % (F2 / 4);
            int col = q * 4;                       // 392 % 4 == 0: channel-uniform
            int c   = (col >= SPL1) ? 1 : 0;
            int k   = col - c * SPL1;
            const float4* src = (const float4*)(xt + (size_t)m * F2) + q;
            float4 v = *src;
            // speculative prefetch of next timestep's tile (same lane address
            // + one-T stride); dropped harmlessly past the end of x.
            __builtin_prefetch((const char*)src + (size_t)BATCH * F2 * 4, 0, 0);
            __bf16* d = &lx[c][m][k];
            d[0] = (__bf16)v.x; d[1] = (__bf16)v.y;
            d[2] = (__bf16)v.z; d[3] = (__bf16)v.w;
        }
        __syncthreads();

        // ---- hidden layer: pipelined 16x16x32 bf16 WMMA, K = 13 x 32 ----
        for (int it = 0; it < ntile; ++it) {
            int tt = w + 8 * it;
            int c = tt / KT_H, nt = tt % KT_H;
            // B base: global, provenance = kernel arg -> global_load_b128
            const __bf16* bp =
                Wb + ((size_t)c * KT_H * NPAD_H + nt * 16 + nl) * 32 + (size_t)g * 8;
            // A base element within the lane's LDS row
            const int a0 = g * 8;   // + kt*32 per k-step
            Frag A[2], B[2];
            A[0].u[0] = *(const uint4*)&lx[c][nl][a0];
            A[0].u[1] = *(const uint4*)&lx[c][nl][a0 + 16];
            B[0].u[0] = *(const uint4*)(bp);
            B[0].u[1] = *(const uint4*)(bp + 16);
            v8f acc0, acc1;
            for (int r = 0; r < 8; ++r) { acc0[r] = 0.f; acc1[r] = 0.f; }
            #pragma unroll
            for (int kt = 0; kt < KT_H; ++kt) {
                const int cur = kt & 1, nxt = cur ^ 1;
                if (kt + 1 < KT_H) {      // issue next k-step's loads first
                    A[nxt].u[0] = *(const uint4*)&lx[c][nl][a0 + (kt + 1) * 32];
                    A[nxt].u[1] = *(const uint4*)&lx[c][nl][a0 + (kt + 1) * 32 + 16];
                    B[nxt].u[0] = *(const uint4*)(bp + (size_t)(kt + 1) * (NPAD_H * 32));
                    B[nxt].u[1] = *(const uint4*)(bp + (size_t)(kt + 1) * (NPAD_H * 32) + 16);
                }
                if (kt & 1)               // two independent accumulator chains
                    acc1 = __builtin_amdgcn_wmma_f32_16x16x32_bf16(
                        false, A[cur].v, false, B[cur].v, (short)0, acc1, false, false);
                else
                    acc0 = __builtin_amdgcn_wmma_f32_16x16x32_bf16(
                        false, A[cur].v, false, B[cur].v, (short)0, acc0, false, false);
            }
            // LIF dynamics on accumulator fragment (m = r + 8*g, n = nt*16+nl)
            for (int r = 0; r < 8; ++r) {
                float cur_h = acc0[r] + acc1[r] + bhr[it];
                float vd    = vh[it][r] + DVM * (ih[it][r] - vh[it][r]);
                float z     = (vd - 1.0f) > 0.f ? 1.f : 0.f;
                vh[it][r]   = (1.f - z) * vd;
                ih[it][r]   = DVS * ih[it][r] + cur_h;
                int m = r + 8 * g;
                lz[c][m][nt * 16 + nl] = (__bf16)z;
            }
        }
        __syncthreads();

        // ---- output layer: waves 0..3, one 16x16 tile, K = 2 x 32 ----
        if (w < OC) {
            const __bf16* op =
                Wob + ((size_t)w * KT_O * NPAD_O + nl) * 32 + (size_t)g * 8;
            Frag A0, A1, B0, B1;
            B0.u[0] = *(const uint4*)(op);
            B0.u[1] = *(const uint4*)(op + 16);
            B1.u[0] = *(const uint4*)(op + NPAD_O * 32);
            B1.u[1] = *(const uint4*)(op + NPAD_O * 32 + 16);
            #pragma unroll
            for (int e = 0; e < 16; ++e) {
                int kb = (e < 8) ? (g * 8 + e) : (16 + g * 8 + (e - 8));
                // kt = 0 chunk (k = kb <= 31 < 50, always valid)
                {
                    int h = w * SPL2 + kb;
                    float s = (float)lz[0][nl][h] + (float)lz[1][nl][h];
                    A0.v[e] = (__bf16)s;
                }
                // kt = 1 chunk (k in [32,64) -> valid only k<50)
                {
                    int k = 32 + kb;
                    float s = 0.f;
                    if (k < SPL2) {
                        int h = w * SPL2 + k;
                        s = (float)lz[0][nl][h] + (float)lz[1][nl][h];
                    }
                    A1.v[e] = (__bf16)s;
                }
            }
            v8f acc0, acc1;
            for (int r = 0; r < 8; ++r) { acc0[r] = 0.f; acc1[r] = 0.f; }
            acc0 = __builtin_amdgcn_wmma_f32_16x16x32_bf16(
                false, A0.v, false, B0.v, (short)0, acc0, false, false);
            acc1 = __builtin_amdgcn_wmma_f32_16x16x32_bf16(
                false, A1.v, false, B1.v, (short)0, acc1, false, false);
            // LI dynamics; output voltage uses vo_new (pre io update)
            for (int r = 0; r < 8; ++r) {
                float cur_o = acc0[r] + acc1[r] + bor;
                float vn    = vo[r] + DVM * (io[r] - vo[r]);
                io[r] = DVS * io[r] + cur_o;
                vo[r] = vn;
                int m = r + 8 * g;
                lvo[w][m][nl] = vn;
            }
        }
        __syncthreads();

        // ---- reduce over OC channels and store out[t] ----
        if (w == 0 && nl < NOUT) {
            for (int r = 0; r < 8; ++r) {
                int m = r + 8 * g;
                float s = lvo[0][m][nl] + lvo[1][m][nl] + lvo[2][m][nl] + lvo[3][m][nl];
                out[((size_t)t * BATCH + b0 + m) * NOUT + nl] = s;
            }
        }
        // no extra barrier needed: lvo is next written only after the next
        // iteration's second barrier; lx/lz writers sit behind barrier #1.
    }
}

extern "C" void kernel_launch(void* const* d_in, const int* in_sizes, int n_in,
                              void* d_out, int out_size, void* d_ws, size_t ws_size,
                              hipStream_t stream) {
    const float* x  = (const float*)d_in[0];   // [200,256,28,28]
    const float* Wh = (const float*)d_in[1];   // [2,200,392]
    const float* bh = (const float*)d_in[2];   // [2,200]
    const float* Wo = (const float*)d_in[3];   // [4,10,50]
    const float* bo = (const float*)d_in[4];   // [4,10]

    __bf16* Wb  = (__bf16*)d_ws;                             // 346,112 B
    __bf16* Wob = (__bf16*)((char*)d_ws + 346112);           //   8,192 B

    const int TOT_H = HC * KT_H * NPAD_H * 32;               // 173,056
    const int TOT_O = OC * KT_O * NPAD_O * 32;               //   4,096
    prep_wh_kernel<<<(TOT_H + 255) / 256, 256, 0, stream>>>(Wh, Wb);
    prep_wo_kernel<<<(TOT_O + 255) / 256, 256, 0, stream>>>(Wo, Wob);

    dendnet_kernel<<<BATCH / MB, 256, 0, stream>>>(x, bh, bo, Wb, Wob, (float*)d_out);
}
